// ExpertModelLSTM_55576876810722
// MI455X (gfx1250) — matrix operand order repaired and checked by reading, not compiled, not run on hardware
//
#include <hip/hip_runtime.h>
#include <hip/hip_bf16.h>

typedef __attribute__((ext_vector_type(16))) _Float16 v16h;
typedef __attribute__((ext_vector_type(2)))  _Float16 h2_t;
typedef __attribute__((ext_vector_type(8)))  float    v8f;
typedef __attribute__((ext_vector_type(4)))  float    v4f;

#define TSEQ 256
#define FIN  64
#define HU   20
#define GW   80   // 4*H gate width
#define NT   5    // 5 N-tiles of 16 covering 80 gate columns
#define MPAD 20   // padded M stride (floats) of transposed z buffer (bank-spread)

// Fast sigmoid: v_mul (fused -log2e scale) + v_exp_f32 + v_add + raw v_rcp_f32.
// No IEEE-division refinement chain on the recurrence critical path.
__device__ __forceinline__ float sigf(float x) {
    return __builtin_amdgcn_rcpf(1.0f + __expf(-x));
}

__device__ __forceinline__ h2_t pkh(float a, float b) {
    auto p = __builtin_amdgcn_cvt_pkrtz(a, b);   // v_cvt_pk_rtz_f16_f32
    h2_t r; r[0] = (_Float16)p[0]; r[1] = (_Float16)p[1];
    return r;
}

__device__ __forceinline__ v8f wmma_f16(v16h a, v16h b, v8f c) {
    // D = A(16x32 f16) * B(32x16 f16) + C(16x16 f32)
    return __builtin_amdgcn_wmma_f32_16x16x32_f16(false, a, false, b, (short)0, c,
                                                  false, false);
}

// Split workgroup barrier usable from wave-divergent code (both waves call it
// the same number of times). Release/acquire fences order LDS (dscnt) around it.
__device__ __forceinline__ void barrier_ws() {
    __builtin_amdgcn_fence(__ATOMIC_RELEASE, "workgroup");
    asm volatile("s_barrier_signal -1\n\t"
                 "s_barrier_wait -1" ::: "memory");
    __builtin_amdgcn_fence(__ATOMIC_ACQUIRE, "workgroup");
}

// B-operand fragment (32x16 f16) from row-major f32 weight W[k][GW].
// Lane layout: N = n0 + lane%16 ; slot s holds K = k0 + (lane/16)*8 + (s&7) + (s>>3)*16.
// Rows k >= krows are zero-padded (H=20 padded to K=32).
__device__ __forceinline__ v16h load_bfrag(const float* __restrict__ W, int krows,
                                           int k0, int n0, int lane) {
    int n  = n0 + (lane & 15);
    int kb = k0 + ((lane >> 4) << 3);
    v16h r;
#pragma unroll
    for (int s = 0; s < 16; ++s) {
        int k  = kb + (s & 7) + ((s >> 3) << 4);
        int kk = (k < krows) ? k : (krows - 1);   // clamp: never read OOB
        float v = W[kk * GW + n];
        r[s] = (_Float16)((k < krows) ? v : 0.0f);
    }
    return r;
}

// A-operand fragment (16x32 f16) from LDS row-major [16][32] halves.
// Lane: M = lane%16; slots 0..7 -> K = hl*8+s ; slots 8..15 -> K = 16+hl*8+(s-8).
__device__ __forceinline__ v16h load_hfrag(const _Float16* hs, int lane) {
    int m  = lane & 15;
    int hl = lane >> 4;
    const _Float16* base = hs + m * 32 + hl * 8;
    union { uint4 q[2]; v16h h; } u;
    u.q[0] = *(const uint4*)(base);        // 16B: K = hl*8 .. hl*8+7
    u.q[1] = *(const uint4*)(base + 16);   // 16B: K = 16+hl*8 ..
    return u.h;
}

// Raw f32 loads of one x row-slice (64 floats across the wave's A layout):
// 8 x b128 per lane, coalescable, issued early for prefetch.
__device__ __forceinline__ void load_xrow(const float* __restrict__ xrow, int hl,
                                          v4f o[8]) {
#pragma unroll
    for (int kc = 0; kc < 2; ++kc)
#pragma unroll
        for (int rr = 0; rr < 2; ++rr)
#pragma unroll
            for (int q = 0; q < 2; ++q)
                o[kc * 4 + rr * 2 + q] =
                    *(const v4f*)(xrow + kc * 32 + rr * 16 + hl * 8 + q * 4);
}

__device__ __forceinline__ v16h cvt_xfrag(const v4f* xr, int kc) {
    v16h r;
#pragma unroll
    for (int p = 0; p < 8; ++p) {        // 8 packed converts per fragment
        int s0 = 2 * p;
        int rr0 = s0 >> 3, e0 = s0 & 7;
        int s1 = s0 + 1;
        int rr1 = s1 >> 3, e1 = s1 & 7;
        float a = xr[kc * 4 + rr0 * 2 + (e0 >> 2)][e0 & 3];
        float b = xr[kc * 4 + rr1 * 2 + (e1 >> 2)][e1 & 3];
        h2_t h = pkh(a, b);
        r[s0] = h[0];
        r[s1] = h[1];
    }
    return r;
}

__global__ __launch_bounds__(64)
void lstm2_fused_kernel(const float* __restrict__ x,
                        const float* __restrict__ W1, const float* __restrict__ U1,
                        const float* __restrict__ b1,
                        const float* __restrict__ W2, const float* __restrict__ U2,
                        const float* __restrict__ b2,
                        const float* __restrict__ Wd, const float* __restrict__ bd,
                        float* __restrict__ out) {
    // LDS: transposed z staging (per layer, bank-padded), double-buffered h1,
    // and h2 state. All h buffers K-padded to 32 halves (pad stays zero).
    __shared__ alignas(16) float     zT0[GW * MPAD];
    __shared__ alignas(16) float     zT1[GW * MPAD];
    __shared__ alignas(16) _Float16  h1buf[2][16 * 32];
    __shared__ alignas(16) _Float16  h2s[16 * 32];

    const int lane  = threadIdx.x & 31;
    const int wid   = threadIdx.x >> 5;
    const int m     = lane & 15;
    const int hl    = lane >> 4;
    const int u0    = hl * 10;          // elementwise unit range per lane
    const int bRow0 = blockIdx.x * 16;

    for (int i = threadIdx.x; i < 16 * 32; i += 64) {
        h1buf[0][i] = (_Float16)0.0f;
        h1buf[1][i] = (_Float16)0.0f;
        h2s[i]      = (_Float16)0.0f;
    }
    __syncthreads();

    if (wid == 0) {
        // ---------------- wave 0: layer-1 LSTM producer ----------------
        v16h W1f[2][NT], U1f[NT];
        float b1v[NT];
#pragma unroll
        for (int tl = 0; tl < NT; ++tl) {
            b1v[tl]    = b1[tl * 16 + m];
            U1f[tl]    = load_bfrag(U1, HU, 0, tl * 16, lane);
            W1f[0][tl] = load_bfrag(W1, FIN, 0,  tl * 16, lane);
            W1f[1][tl] = load_bfrag(W1, FIN, 32, tl * 16, lane);
        }
        float c1[10];
#pragma unroll
        for (int j = 0; j < 10; ++j) c1[j] = 0.0f;

        const float* xrow = x + (size_t)(bRow0 + m) * TSEQ * FIN;
        v4f xc[8], xn[8];
        load_xrow(xrow, hl, xc);

        for (int t = 0; t < TSEQ; ++t) {
            // prefetch x[t+1] (independent of the recurrence chain)
            int tn = (t + 1 < TSEQ) ? (t + 1) : t;
            load_xrow(xrow + (size_t)tn * FIN, hl, xn);

            v16h hA  = load_hfrag(h1buf[(t & 1) ^ 1], lane);  // h1[t-1]
            v16h xA0 = cvt_xfrag(xc, 0);
            v16h xA1 = cvt_xfrag(xc, 1);

#pragma unroll
            for (int tl = 0; tl < NT; ++tl) {
                v8f acc;
#pragma unroll
                for (int e = 0; e < 8; ++e) acc[e] = b1v[tl];   // bias (hoisted)
                acc = wmma_f16(hA,  U1f[tl],    acc);           // h U1
                acc = wmma_f16(xA0, W1f[0][tl], acc);           // x W1 (K 0..31)
                acc = wmma_f16(xA1, W1f[1][tl], acc);           // x W1 (K 32..63)
                // store transposed: zT[N][M], lane's 8 M-contiguous values
                float* p = &zT0[(tl * 16 + m) * MPAD + hl * 8];
                v4f lo, hi;
#pragma unroll
                for (int e = 0; e < 4; ++e) { lo[e] = acc[e]; hi[e] = acc[4 + e]; }
                *(v4f*)p       = lo;
                *(v4f*)(p + 4) = hi;
            }

            // gate nonlinearity + cell update: lane owns (M=m, u=u0..u0+9)
            float hv[10];
#pragma unroll
            for (int j = 0; j < 10; ++j) {
                int u   = u0 + j;
                float iv = sigf(zT0[(u     ) * MPAD + m]);
                float fv = sigf(zT0[(u + 20) * MPAD + m]);
                float gv =       zT0[(u + 40) * MPAD + m];
                float ov = sigf(zT0[(u + 60) * MPAD + m]);
                gv = gv > 0.0f ? gv : 0.0f;                  // relu(g)
                c1[j] = fv * c1[j] + iv * gv;
                float cr = c1[j] > 0.0f ? c1[j] : 0.0f;      // relu(c)
                hv[j] = ov * cr;
            }
            _Float16* hout = h1buf[t & 1];
#pragma unroll
            for (int p = 0; p < 5; ++p)                       // 5 packed b32 stores
                *(h2_t*)(hout + m * 32 + u0 + 2 * p) = pkh(hv[2 * p], hv[2 * p + 1]);

#pragma unroll
            for (int i = 0; i < 8; ++i) xc[i] = xn[i];
            barrier_ws();                                    // publish h1[t]
        }
        barrier_ws();  // drain: match consumer's trailing rendezvous
    } else {
        // ---------------- wave 1: layer-2 LSTM consumer + dense ----------------
        v16h W2f[NT], U2f[NT];
        float b2v[NT];
#pragma unroll
        for (int tl = 0; tl < NT; ++tl) {
            b2v[tl] = b2[tl * 16 + m];
            W2f[tl] = load_bfrag(W2, HU, 0, tl * 16, lane);
            U2f[tl] = load_bfrag(U2, HU, 0, tl * 16, lane);
        }
        float c2[10], hv2[10], wdv[10];
#pragma unroll
        for (int j = 0; j < 10; ++j) {
            c2[j]  = 0.0f;
            hv2[j] = 0.0f;
            wdv[j] = Wd[u0 + j];
        }

        barrier_ws();  // wait for h1[0]
        for (int t = 0; t < TSEQ; ++t) {
            v16h h1A = load_hfrag(h1buf[t & 1], lane);   // h1[t] (just published)
            v16h h2A = load_hfrag(h2s, lane);            // h2[t-1]

#pragma unroll
            for (int tl = 0; tl < NT; ++tl) {
                v8f acc;
#pragma unroll
                for (int e = 0; e < 8; ++e) acc[e] = b2v[tl];
                acc = wmma_f16(h1A, W2f[tl], acc);
                acc = wmma_f16(h2A, U2f[tl], acc);
                float* p = &zT1[(tl * 16 + m) * MPAD + hl * 8];
                v4f lo, hi;
#pragma unroll
                for (int e = 0; e < 4; ++e) { lo[e] = acc[e]; hi[e] = acc[4 + e]; }
                *(v4f*)p       = lo;
                *(v4f*)(p + 4) = hi;
            }

#pragma unroll
            for (int j = 0; j < 10; ++j) {
                int u   = u0 + j;
                float iv = sigf(zT1[(u     ) * MPAD + m]);
                float fv = sigf(zT1[(u + 20) * MPAD + m]);
                float gv =       zT1[(u + 40) * MPAD + m];
                float ov = sigf(zT1[(u + 60) * MPAD + m]);
                gv = gv > 0.0f ? gv : 0.0f;
                c2[j] = fv * c2[j] + iv * gv;
                float cr = c2[j] > 0.0f ? c2[j] : 0.0f;
                hv2[j] = ov * cr;
            }
#pragma unroll
            for (int p = 0; p < 5; ++p)
                *(h2_t*)(h2s + m * 32 + u0 + 2 * p) = pkh(hv2[2 * p], hv2[2 * p + 1]);
            barrier_ws();
        }

        // dense head: out[M] = sum_u h2[M][u] * Wd[u] + bd
        float part = 0.0f;
#pragma unroll
        for (int j = 0; j < 10; ++j) part += hv2[j] * wdv[j];
        part += __shfl_xor(part, 16, 32);   // combine u 0..9 with u 10..19
        if (lane < 16) out[bRow0 + m] = part + bd[0];
    }
}

extern "C" void kernel_launch(void* const* d_in, const int* in_sizes, int n_in,
                              void* d_out, int out_size, void* d_ws, size_t ws_size,
                              hipStream_t stream) {
    (void)n_in; (void)out_size; (void)d_ws; (void)ws_size;
    const float* x  = (const float*)d_in[0];
    const float* W1 = (const float*)d_in[1];
    const float* U1 = (const float*)d_in[2];
    const float* b1 = (const float*)d_in[3];
    const float* W2 = (const float*)d_in[4];
    const float* U2 = (const float*)d_in[5];
    const float* b2 = (const float*)d_in[6];
    const float* Wd = (const float*)d_in[7];
    const float* bd = (const float*)d_in[8];
    float* out = (float*)d_out;

    int B = in_sizes[0] / (TSEQ * FIN);   // 2048
    int blocks = B / 16;                  // 128 workgroups x 2 waves
    lstm2_fused_kernel<<<dim3(blocks), dim3(64), 0, stream>>>(
        x, W1, U1, b1, W2, U2, b2, Wd, bd, out);
}
